// GPNN_44813688767306
// MI455X (gfx1250) — matrix operand order repaired and checked by reading, not compiled, not run on hardware
//
#include <hip/hip_runtime.h>
#include <hip/hip_bf16.h>
#include <math.h>

typedef _Float16 v16h __attribute__((ext_vector_type(16)));
typedef _Float16 v8h  __attribute__((ext_vector_type(8)));
typedef float    v8f  __attribute__((ext_vector_type(8)));

#define DV 512
#define DE 256
#define NN 128
#define BB 8
#define KK 3
#define CC 117
#define DCOMB 1280   // 2*DV + DE
#define GATES 1536   // 3*DV

__device__ __forceinline__ v16h combine16(v8h lo, v8h hi) {
  return __builtin_shufflevector(lo, hi, 0,1,2,3,4,5,6,7,8,9,10,11,12,13,14,15);
}

__device__ __forceinline__ v16h ld16(const _Float16* p) {
  return combine16(*(const v8h*)p, *(const v8h*)(p + 8));
}

__device__ __forceinline__ float sigmoidf_(float x) { return 1.0f / (1.0f + expf(-x)); }

// ---------------------------------------------------------------------------
// Prep: dtype casts / transposes into the layouts the hot loop wants.
// ---------------------------------------------------------------------------
__global__ void prep_kernel(const float* __restrict__ ef, const float* __restrict__ nf,
                            const float* __restrict__ wmsg,
                            const float* __restrict__ w_ih, const float* __restrict__ w_hh,
                            _Float16* __restrict__ E16, _Float16* __restrict__ HT16,
                            float* __restrict__ Hf32, _Float16* __restrict__ wmsg16,
                            _Float16* __restrict__ wihT, _Float16* __restrict__ whhT) {
  size_t tid = (size_t)blockIdx.x * blockDim.x + threadIdx.x;
  size_t stride = (size_t)gridDim.x * blockDim.x;
  for (size_t i = tid; i < (size_t)BB*NN*NN*DE; i += stride) E16[i] = (_Float16)ef[i];
  for (size_t i = tid; i < (size_t)BB*NN*DV; i += stride) {
    float v = nf[i]; Hf32[i] = v; HT16[i] = (_Float16)v;
  }
  for (size_t i = tid; i < (size_t)DE*DCOMB; i += stride) wmsg16[i] = (_Float16)wmsg[i];
  for (size_t i = tid; i < (size_t)GATES*DE; i += stride) {
    size_t row = i / DE, k = i % DE;
    wihT[k*GATES + row] = (_Float16)w_ih[i];
  }
  for (size_t i = tid; i < (size_t)GATES*DV; i += stride) {
    size_t row = i / DV, k = i % DV;
    whhT[k*GATES + row] = (_Float16)w_hh[i];
  }
}

// ---------------------------------------------------------------------------
// Main persistent kernel: one 512-thread (16 wave32) block per batch element.
// Wave w owns output row-tile cm=w; 8 persistent accumulators cover all 8
// column tiles so consecutive WMMAs are independent (kt-outer / cn-inner).
// ---------------------------------------------------------------------------
__global__ __launch_bounds__(512, 1)
void gpnn_main(const float* __restrict__ w_link, const float* __restrict__ b_link,
               const float* __restrict__ b_msg,
               const _Float16* __restrict__ wihT, const _Float16* __restrict__ whhT,
               const float* __restrict__ b_ih, const float* __restrict__ b_hh,
               const _Float16* __restrict__ wmsg16,
               _Float16* __restrict__ HT16g, float* __restrict__ Hf32g,
               _Float16* __restrict__ E16g, float* __restrict__ out) {
  __shared__ __align__(16) float    s_wl[DCOMB];
  __shared__ __align__(16) float    s_bm[DE];
  __shared__ __align__(16) float    s_u[NN];
  __shared__ __align__(16) float    s_v[NN];
  __shared__ __align__(16) float    s_hv[DV];
  __shared__ __align__(16) _Float16 s_hv16[DV];
  __shared__ __align__(16) float    s_msum[DE];

  const int b     = blockIdx.x;
  const int tid   = threadIdx.x;
  const int lane  = tid & 31;
  const int w     = tid >> 5;      // wave id 0..15 -> output row tile cm
  const int lhalf = lane >> 4;     // 0/1 half-wave
  const int l15   = lane & 15;

  _Float16* HT16 = HT16g + (size_t)b*NN*DV;
  float*    Hf32 = Hf32g + (size_t)b*NN*DV;
  _Float16* E16  = E16g  + (size_t)b*NN*NN*DE;
  float*    adjO = out   + (size_t)b*NN*NN;   // live adj buffer == output slice

  for (int idx = tid; idx < DCOMB; idx += 512) s_wl[idx] = w_link[idx];
  for (int idx = tid; idx < DE;    idx += 512) s_bm[idx] = b_msg[idx];
  const float bl = b_link[0];
  __syncthreads();

  const int arow = w*16 + l15;                         // A row (M), per ISA layout
  const int c0   = w*16 + lhalf*8;                     // D layout: 8 consecutive c/lane
  const _Float16* abase = wmsg16 + (size_t)arow*DCOMB + lhalf*16;
  const _Float16* hbase = HT16 + (size_t)l15*DV + lhalf*16;   // + cn*16*DV + kt*32

  for (int kit = 0; kit < KK; ++kit) {
    // ---------------- adjacency: adj = sigmoid(u_i + v_j + wl_e.e + b) -----
    if (tid < NN) {
      const float* hr = Hf32 + (size_t)tid*DV;
      float su = 0.f, sv = 0.f;
      for (int d = 0; d < DV; ++d) { float hv = hr[d]; su += s_wl[DV+d]*hv; sv += s_wl[d]*hv; }
      s_u[tid] = su; s_v[tid] = sv;
    }
    __syncthreads();
    for (int idx = tid; idx < NN*NN; idx += 512) {
      const int i = idx >> 7, j = idx & (NN-1);
      const v8h* ep = (const v8h*)(E16 + ((size_t)i*NN + j)*DE);
      float s = 0.f;
      for (int cc = 0; cc < DE/8; ++cc) {
        v8h ev = ep[cc];
        #pragma unroll
        for (int q = 0; q < 8; ++q) s += s_wl[2*DV + cc*8 + q] * (float)ev[q];
      }
      adjO[idx] = sigmoidf_(s_u[i] + s_v[j] + s + bl);
    }
    __threadfence();
    __syncthreads();

    // ---------------- sequential node scan --------------------------------
    for (int i = 0; i < NN; ++i) {
      // stage h_v (column i of h)
      { float hv = Hf32[(size_t)i*DV + tid]; s_hv[tid] = hv; s_hv16[tid] = (_Float16)hv; }
      __syncthreads();

      v8f acc[8];
      #pragma unroll
      for (int cn = 0; cn < 8; ++cn)
        #pragma unroll
        for (int q = 0; q < 8; ++q) acc[cn][q] = 0.f;

      // one base + immediate offsets for all E loads: col cn is +cn*16*DE halfs
      const _Float16* ebase = E16 + ((size_t)i*NN + l15)*DE + lhalf*16;

      // --- E segment (k 1024..1279): old e_row; 1 A load + 8 WMMA per kt
      #pragma unroll 2
      for (int kt = 0; kt < 8; ++kt) {
        v16h a = ld16(abase + (2*DV + kt*32));
        #pragma unroll
        for (int cn = 0; cn < 8; ++cn) {
          v16h bm = ld16(ebase + cn*(16*DE) + kt*32);
          acc[cn] = __builtin_amdgcn_wmma_f32_16x16x32_f16(false, a, false, bm, (short)0,
                                                           acc[cn], false, false);
        }
      }
      __syncthreads();   // ALL e_row reads in the block complete before any write

      // --- h_v broadcast segment (k 0..511): B shared by all 8 column tiles
      #pragma unroll 2
      for (int kt = 0; kt < 16; ++kt) {
        v16h a  = ld16(abase + kt*32);
        v16h bm = ld16(&s_hv16[kt*32 + lhalf*16]);
        #pragma unroll
        for (int cn = 0; cn < 8; ++cn)
          acc[cn] = __builtin_amdgcn_wmma_f32_16x16x32_f16(false, a, false, bm, (short)0,
                                                           acc[cn], false, false);
      }
      // --- h segment (k 512..1023): node-major f16 h; col cn is +cn*16*DV halfs
      #pragma unroll 2
      for (int kt = 0; kt < 16; ++kt) {
        v16h a = ld16(abase + (DV + kt*32));
        #pragma unroll
        for (int cn = 0; cn < 8; ++cn) {
          v16h bm = ld16(hbase + cn*(16*DV) + kt*32);
          acc[cn] = __builtin_amdgcn_wmma_f32_16x16x32_f16(false, a, false, bm, (short)0,
                                                           acc[cn], false, false);
        }
      }

      // --- tail: bias + relu, write m into e-row, accumulate msum partials
      float psum[8];
      #pragma unroll
      for (int q = 0; q < 8; ++q) psum[q] = 0.f;
      _Float16* estore = E16 + ((size_t)i*NN + l15)*DE + c0;
      #pragma unroll
      for (int cn = 0; cn < 8; ++cn) {
        const float adjv = adjO[i*NN + cn*16 + l15];
        v8h mh;
        #pragma unroll
        for (int q = 0; q < 8; ++q) {
          float f = acc[cn][q] + s_bm[c0 + q];
          f = f > 0.f ? f : 0.f;
          mh[q] = (_Float16)f;
          psum[q] += f * adjv;
        }
        *(v8h*)(estore + cn*(16*DE)) = mh;
      }

      // deterministic msum reduction: 16-lane halves -> unique LDS slots
      #pragma unroll
      for (int q = 0; q < 8; ++q) {
        float p = psum[q];
        p += __shfl_xor(p, 1, 32);
        p += __shfl_xor(p, 2, 32);
        p += __shfl_xor(p, 4, 32);
        p += __shfl_xor(p, 8, 32);
        psum[q] = p;
      }
      if (l15 == 0) {
        #pragma unroll
        for (int q = 0; q < 8; ++q) s_msum[c0 + q] = psum[q];
      }
      __syncthreads();

      // ---- GRU cell: thread tid owns hidden dim d = tid -------------------
      {
        const int d = tid;
        float gir = b_ih[d], giz = b_ih[DV + d], gin = b_ih[2*DV + d];
        for (int k2 = 0; k2 < DE; ++k2) {                // coalesced f16 rows
          float mv = s_msum[k2];
          const _Float16* wr = wihT + (size_t)k2*GATES;
          gir += (float)wr[d]*mv; giz += (float)wr[DV+d]*mv; gin += (float)wr[2*DV+d]*mv;
        }
        float ghr = b_hh[d], ghz = b_hh[DV + d], ghn = b_hh[2*DV + d];
        for (int k2 = 0; k2 < DV; ++k2) {
          float hv = s_hv[k2];
          const _Float16* wr = whhT + (size_t)k2*GATES;
          ghr += (float)wr[d]*hv; ghz += (float)wr[DV+d]*hv; ghn += (float)wr[2*DV+d]*hv;
        }
        float r  = sigmoidf_(gir + ghr);
        float z  = sigmoidf_(giz + ghz);
        float nn = tanhf(gin + r*ghn);
        float hnew = (1.f - z)*nn + z*s_hv[d];
        Hf32[(size_t)i*DV + d] = hnew;
        HT16[(size_t)i*DV + d] = (_Float16)hnew;
      }
      __threadfence();
      __syncthreads();
    } // node i
  } // kit
}

// ---------------------------------------------------------------------------
// Readout: pred_node_labels[b][n][c] = w_ro[c] . h[:,n] + b_ro[c]
// ---------------------------------------------------------------------------
__global__ __launch_bounds__(128)
void readout_kernel(const float* __restrict__ w_ro, const float* __restrict__ b_ro,
                    const float* __restrict__ Hf32g, float* __restrict__ out) {
  __shared__ float sh[DV];
  const int bn  = blockIdx.x;   // b*NN + n
  const int tid = threadIdx.x;
  const float* hrow = Hf32g + (size_t)bn*DV;
  for (int k = tid; k < DV; k += 128) sh[k] = hrow[k];
  __syncthreads();
  if (tid < CC) {
    const float* wr = w_ro + (size_t)tid*DV;
    float s = b_ro[tid];
    for (int d = 0; d < DV; ++d) s += wr[d]*sh[d];
    out[(size_t)BB*NN*NN + (size_t)bn*CC + tid] = s;
  }
}

// ---------------------------------------------------------------------------
extern "C" void kernel_launch(void* const* d_in, const int* in_sizes, int n_in,
                              void* d_out, int out_size, void* d_ws, size_t ws_size,
                              hipStream_t stream) {
  const float* ef     = (const float*)d_in[0];   // edge_features (B,N,N,DE)
  const float* nf     = (const float*)d_in[1];   // node_features (B,N,DV)
  const float* w_link = (const float*)d_in[6];
  const float* b_link = (const float*)d_in[7];
  const float* w_msg  = (const float*)d_in[8];
  const float* b_msg  = (const float*)d_in[9];
  const float* w_ih   = (const float*)d_in[10];
  const float* w_hh   = (const float*)d_in[11];
  const float* b_ih   = (const float*)d_in[12];
  const float* b_hh   = (const float*)d_in[13];
  const float* w_ro   = (const float*)d_in[14];
  const float* b_ro   = (const float*)d_in[15];

  char* ws = (char*)d_ws;
  _Float16* wmsg16 = (_Float16*)(ws);                 //    655,360 B
  _Float16* HT16   = (_Float16*)(ws + 655360);        //  1,048,576 B
  float*    Hf32   = (float*)   (ws + 1703936);       //  2,097,152 B
  _Float16* E16    = (_Float16*)(ws + 3801088);       // 67,108,864 B
  _Float16* wihT   = (_Float16*)(ws + 70909952);      //    786,432 B
  _Float16* whhT   = (_Float16*)(ws + 71696384);      //  1,572,864 B  (~73.3 MB total)

  float* out = (float*)d_out;

  prep_kernel<<<4096, 256, 0, stream>>>(ef, nf, w_msg, w_ih, w_hh,
                                        E16, HT16, Hf32, wmsg16, wihT, whhT);
  gpnn_main<<<BB, 512, 0, stream>>>(w_link, b_link, b_msg, wihT, whhT, b_ih, b_hh,
                                    wmsg16, HT16, Hf32, E16, out);
  readout_kernel<<<BB*NN, 128, 0, stream>>>(w_ro, b_ro, Hf32, out);
}
